// GPTCat_45148696216179
// MI455X (gfx1250) — compile-verified, matrix operand order checked by reading, not run on hardware
//
#include <hip/hip_runtime.h>

// ---------------------------------------------------------------------------
// CDNA5 (gfx1250) decision-transformer forward pass.
// All GEMMs + attention use v_wmma_f32_16x16x32_bf16 (bf16 in, fp32 acc).
// GEMM stages A/B tiles into double-buffered LDS with async copies
// (global_load_async_to_lds_b128 + s_wait_asynccnt) to overlap DMA w/ WMMA.
// Flash-attention: per-wave 16-row Q tile, K/V streamed in 32-wide chunks,
// P tile staged through LDS to go from WMMA C-layout to A-fragment layout.
// ---------------------------------------------------------------------------

typedef unsigned short u16;
typedef unsigned int   u32;
typedef __attribute__((ext_vector_type(16))) __bf16 v16bf;
typedef __attribute__((ext_vector_type(8)))  float  v8f;
typedef __attribute__((ext_vector_type(8)))  u32    v8u;

#define DEV static __device__ __forceinline__

DEV u16 f2bf(float f) {
  u32 x = __builtin_bit_cast(u32, f);
  u32 r = (x + 0x7FFFu + ((x >> 16) & 1u)) >> 16;   // round-to-nearest-even
  return (u16)r;
}

// Load a 16-bit A/B WMMA fragment from a row-major bf16 row pointer.
// Lane layout (ISA 7.12.2): lane L -> row = L&15, khi = (L>>4)*8,
// elements = two contiguous 16B chunks at k offsets khi and khi+16.
DEV v16bf frag16(const u16* rowp, int khi) {
  union { v8u u; v16bf v; } r;
  uint4 a = *reinterpret_cast<const uint4*>(rowp + khi);
  uint4 b = *reinterpret_cast<const uint4*>(rowp + khi + 16);
  r.u[0] = a.x; r.u[1] = a.y; r.u[2] = a.z; r.u[3] = a.w;
  r.u[4] = b.x; r.u[5] = b.y; r.u[6] = b.z; r.u[7] = b.w;
  return r.v;
}

DEV v8f wmma_bf(v16bf a, v16bf b, v8f c) {
  return __builtin_amdgcn_wmma_f32_16x16x32_bf16(false, a, false, b, (short)0, c,
                                                 false, false);
}

// Async copy 16B global -> LDS (ASYNCcnt-tracked DMA path).
// Generic pointers to __shared__ have the LDS byte offset in addr[31:0]
// (ISA 10.2 aperture mapping), so truncation yields the VDST LDS address.
DEV void async_b128(const u16* shared_dst, const u16* gsrc) {
  u32 lds = (u32)(size_t)shared_dst;
  unsigned long long ga = (unsigned long long)(size_t)gsrc;
  asm volatile("global_load_async_to_lds_b128 %0, %1, off"
               :: "v"(lds), "v"(ga) : "memory");
}

DEV float red_max16(float v) {
  for (int m = 1; m < 16; m <<= 1) v = fmaxf(v, __shfl_xor(v, m, 16));
  return v;
}
DEV float red_sum16(float v) {
  for (int m = 1; m < 16; m <<= 1) v += __shfl_xor(v, m, 16);
  return v;
}

// ---------------------------------------------------------------------------
// WMMA GEMM: C[M,N] = act(A[M,K] * Bt[N,K]^T + bias) (+ resid)
// A, Bt bf16; acc fp32; out fp32 or bf16. Block = 256 thr = 8 waves.
// Block tile 128x64, wave tile 32x32 (2x2 of 16x16). K staged in 64-wide
// chunks through double-buffered LDS filled by async b128 copies:
//   A tile 128x64 = 16KB (4 async instr/block), B tile 64x64 = 8KB (2 instr).
// Requires K % 64 == 0 (all K here: 64/256/768/3072).
// ---------------------------------------------------------------------------
template <int ACT, bool OUT_BF16>
__global__ __launch_bounds__(256) void gemm_wmma(
    const u16* __restrict__ A, int lda,
    const u16* __restrict__ Bt, int ldb, int Np,
    const float* __restrict__ bias,
    const float* __restrict__ resid,
    void* __restrict__ Cout, int ldc,
    int M, int N, int K) {
  __shared__ __align__(16) u16 sA[2][128 * 64];
  __shared__ __align__(16) u16 sB[2][64 * 64];

  int tid = threadIdx.x, lane = tid & 31, wid = tid >> 5;
  int wm = wid >> 1, wn = wid & 1;
  int nl = lane & 15, hi = lane >> 4;
  int khi = hi << 3;
  int mblk = blockIdx.y * 128;
  int nblk = blockIdx.x * 64;

  v8f acc[2][2] = {{{0,0,0,0,0,0,0,0}, {0,0,0,0,0,0,0,0}},
                   {{0,0,0,0,0,0,0,0}, {0,0,0,0,0,0,0,0}}};

  // cooperative async stage of one K-chunk (64 cols) into buffer `buf`
  auto stage = [&](int k0, int buf) {
    for (int it = 0; it < 4; it++) {           // A: 1024 chunks of 16B
      int i = tid + 256 * it;
      int row = i >> 3, c = i & 7;
      int m = mblk + row; m = (m < M) ? m : (M - 1);
      async_b128(&sA[buf][row * 64 + c * 8], A + (size_t)m * lda + k0 + c * 8);
    }
    for (int it = 0; it < 2; it++) {           // B: 512 chunks of 16B
      int i = tid + 256 * it;
      int row = i >> 3, c = i & 7;
      int n = nblk + row; n = (n < Np) ? n : (Np - 1);
      async_b128(&sB[buf][row * 64 + c * 8], Bt + (size_t)n * ldb + k0 + c * 8);
    }
  };

  int nstages = K >> 6;
  stage(0, 0);
  for (int s = 0; s < nstages; s++) {
    if (s + 1 < nstages) {
      stage((s + 1) << 6, (s + 1) & 1);
      asm volatile("s_wait_asynccnt 6" ::: "memory");  // stage s complete
    } else {
      asm volatile("s_wait_asynccnt 0" ::: "memory");
    }
    __syncthreads();
    const u16* bufA = sA[s & 1];
    const u16* bufB = sB[s & 1];
    for (int kc = 0; kc < 2; kc++) {
      v16bf a0 = frag16(bufA + (wm * 32 + nl) * 64 + kc * 32, khi);
      v16bf a1 = frag16(bufA + (wm * 32 + 16 + nl) * 64 + kc * 32, khi);
      v16bf b0 = frag16(bufB + (wn * 32 + nl) * 64 + kc * 32, khi);
      v16bf b1 = frag16(bufB + (wn * 32 + 16 + nl) * 64 + kc * 32, khi);
      acc[0][0] = wmma_bf(a0, b0, acc[0][0]);
      acc[0][1] = wmma_bf(a0, b1, acc[0][1]);
      acc[1][0] = wmma_bf(a1, b0, acc[1][0]);
      acc[1][1] = wmma_bf(a1, b1, acc[1][1]);
    }
    __syncthreads();   // all waves done reading before buffer reuse
  }

  for (int ni = 0; ni < 2; ni++) {
    int n = nblk + wn * 32 + ni * 16 + nl;
    if (n >= N) continue;
    float bn = bias ? bias[n] : 0.f;
    for (int mi = 0; mi < 2; mi++) {
      for (int r = 0; r < 8; r++) {
        int m = mblk + wm * 32 + mi * 16 + r + 8 * hi;
        if (m >= M) continue;
        float v = acc[mi][ni][r] + bn;
        if (ACT == 1) v = v > 0.f ? v : 0.f;
        else if (ACT == 2) v = 0.5f * v * (1.f + erff(v * 0.70710678118f));
        if (resid) v += resid[(size_t)m * ldc + n];
        if (OUT_BF16) ((u16*)Cout)[(size_t)m * ldc + n] = f2bf(v);
        else          ((float*)Cout)[(size_t)m * ldc + n] = v;
      }
    }
  }
}

// ---------------------------------------------------------------------------
// Flash attention (causal). Q,K bf16 [8192,768] head-interleaved; Vt bf16
// [B*H][64][512]. Per wave: one 16-row Q tile; loop over 32-wide K/V blocks.
// grid = (4 q-tiles-of-128, B*H); block 256 = 8 waves.
// ---------------------------------------------------------------------------
__global__ __launch_bounds__(256) void attn_flash(
    const u16* __restrict__ Q, const u16* __restrict__ Kb,
    const u16* __restrict__ Vt, u16* __restrict__ Y) {
  __shared__ __align__(16) u16 sP[8][512];  // per-wave 16x32 bf16 P tile
  const float SCALE = 0.125f;               // 1/sqrt(64)
  int lane = threadIdx.x & 31, wid = threadIdx.x >> 5;
  int nl = lane & 15, hi = lane >> 4, khi = hi << 3;
  int bh = blockIdx.y, b = bh / 12, h = bh % 12;
  int q0 = blockIdx.x * 128 + wid * 16;
  int tb = b * 512;

  const u16* qr = Q + (size_t)(tb + q0 + nl) * 768 + h * 64;
  v16bf aQ0 = frag16(qr, khi);
  v16bf aQ1 = frag16(qr + 32, khi);

  v8f o[4] = {{0,0,0,0,0,0,0,0}, {0,0,0,0,0,0,0,0},
              {0,0,0,0,0,0,0,0}, {0,0,0,0,0,0,0,0}};
  float mrow[8], lrow[8];
  for (int r = 0; r < 8; r++) { mrow[r] = -1e30f; lrow[r] = 0.f; }

  for (int j = 0; j < q0 + 16; j += 32) {
    v8f s0 = {0,0,0,0,0,0,0,0}, s1 = {0,0,0,0,0,0,0,0};
    {
      const u16* kr = Kb + (size_t)(tb + j + nl) * 768 + h * 64;
      s0 = wmma_bf(aQ0, frag16(kr, khi), s0);
      s0 = wmma_bf(aQ1, frag16(kr + 32, khi), s0);
    }
    bool t1 = (j < q0);  // uniform per wave
    if (t1) {
      const u16* kr = Kb + (size_t)(tb + j + 16 + nl) * 768 + h * 64;
      s1 = wmma_bf(aQ0, frag16(kr, khi), s1);
      s1 = wmma_bf(aQ1, frag16(kr + 32, khi), s1);
    }
    float facr[8];
    for (int r = 0; r < 8; r++) {
      int row = q0 + r + 8 * hi;
      float v0 = (j + nl <= row) ? s0[r] * SCALE : -1e30f;
      float v1 = (t1 && (j + 16 + nl <= row)) ? s1[r] * SCALE : -1e30f;
      float tmax = red_max16(fmaxf(v0, v1));
      float nm = fmaxf(mrow[r], tmax);
      float fac = __expf(mrow[r] - nm);
      mrow[r] = nm; facr[r] = fac;
      float p0 = __expf(v0 - nm), p1 = __expf(v1 - nm);
      lrow[r] = lrow[r] * fac + red_sum16(p0 + p1);
      sP[wid][(r + 8 * hi) * 32 + nl]      = f2bf(p0);
      sP[wid][(r + 8 * hi) * 32 + nl + 16] = f2bf(p1);
    }
    for (int nt = 0; nt < 4; nt++)
      for (int r = 0; r < 8; r++) o[nt][r] *= facr[r];
    v16bf aP = frag16(&sP[wid][nl * 32], khi);  // ds_load fragments
    for (int nt = 0; nt < 4; nt++) {
      const u16* vr = Vt + ((size_t)(bh * 64 + nt * 16 + nl)) * 512 + j;
      o[nt] = wmma_bf(aP, frag16(vr, khi), o[nt]);
    }
  }
  for (int nt = 0; nt < 4; nt++)
    for (int r = 0; r < 8; r++) {
      int row = tb + q0 + r + 8 * hi;
      int col = h * 64 + nt * 16 + nl;
      float v = o[nt][r] / lrow[r];
      Y[(size_t)row * 768 + col] = f2bf(v);
    }
}

// ---------------------------------------------------------------------------
// LayerNorm: fp32 [rows,768] -> bf16. One wave per row.
// ---------------------------------------------------------------------------
__global__ __launch_bounds__(256) void ln_kernel(
    const float* __restrict__ x, const float* __restrict__ g,
    const float* __restrict__ b, u16* __restrict__ out, int rows) {
  int lane = threadIdx.x & 31, wid = threadIdx.x >> 5;
  int row = blockIdx.x * 8 + wid;
  if (row >= rows) return;
  const float* xr = x + (size_t)row * 768;
  float vals[24], s = 0.f, s2 = 0.f;
  for (int i = 0; i < 24; i++) {
    float v = xr[lane + 32 * i];
    vals[i] = v; s += v; s2 += v * v;
  }
  for (int m = 1; m < 32; m <<= 1) {
    s += __shfl_xor(s, m, 32);
    s2 += __shfl_xor(s2, m, 32);
  }
  float mean = s * (1.f / 768.f);
  float rstd = rsqrtf(s2 * (1.f / 768.f) - mean * mean + 1e-5f);
  u16* orow = out + (size_t)row * 768;
  for (int i = 0; i < 24; i++) {
    int c = lane + 32 * i;
    orow[c] = f2bf((vals[i] - mean) * rstd * g[c] + b[c]);
  }
}

// Weight convert: W fp32 [fi,fo] -> Wt bf16 [Npad,Kpad] (transposed, zero-pad)
__global__ void wconv_kernel(const float* __restrict__ W, u16* __restrict__ Wt,
                             int fi, int fo, int Kpad, int Npad) {
  int i = blockIdx.x * 256 + threadIdx.x;
  if (i >= Npad * Kpad) return;
  int n = i / Kpad, k = i % Kpad;
  Wt[i] = (k < fi && n < fo) ? f2bf(W[(size_t)k * fo + n]) : (u16)0;
}

// states fp32 [4096,50] -> bf16 [4096,64] zero-pad
__global__ void pad_states_kernel(const float* __restrict__ s, u16* __restrict__ o) {
  int i = blockIdx.x * 256 + threadIdx.x;
  if (i >= 4096 * 64) return;
  int r = i >> 6, c = i & 63;
  o[i] = (c < 50) ? f2bf(s[r * 50 + c]) : (u16)0;
}

// concat(actions[:, :255], key_states[:, :255]) -> bf16 [4080,64]
__global__ void concat_act_kernel(const float* __restrict__ a,
                                  const float* __restrict__ k,
                                  u16* __restrict__ o) {
  int i = blockIdx.x * 256 + threadIdx.x;
  if (i >= 4080 * 64) return;
  int r = i >> 6, c = i & 63;
  int b = r / 255, t = r % 255, base = b * 256 + t;
  float v = 0.f;
  if (c < 10) v = a[base * 10 + c];
  else if (c < 60) v = k[base * 50 + (c - 10)];
  o[i] = f2bf(v);
}

// token interleave + positional embeddings -> X fp32 [8192,768]
__global__ void assemble_kernel(const float* __restrict__ se,
                                const float* __restrict__ ae,
                                const float* __restrict__ gpe,
                                const float* __restrict__ lpe,
                                const int* __restrict__ ts,
                                float* __restrict__ X) {
  int i = blockIdx.x * 256 + threadIdx.x;
  if (i >= 8192 * 768) return;
  int tok = i / 768, d = i % 768;
  int b = tok >> 9, slot = tok & 511, s = slot >> 1;
  float v = 0.f;
  if ((slot & 1) == 0) v = se[((size_t)(b * 256 + s)) * 768 + d];
  else if (s < 255)    v = ae[((size_t)(b * 255 + s)) * 768 + d];
  v += gpe[(size_t)ts[b] * 768 + d] + lpe[(size_t)s * 768 + d];
  X[i] = v;
}

// V bf16 [8192,768] -> Vt bf16 [B*H][64][512]
__global__ void vtrans_kernel(const u16* __restrict__ V, u16* __restrict__ Vt) {
  int i = blockIdx.x * 256 + threadIdx.x;
  if (i >= 192 * 64 * 512) return;
  int t = i & 511, dh = (i >> 9) & 63, bh = i >> 15;
  int b = bh / 12, h = bh % 12;
  Vt[i] = V[((size_t)(b * 512 + t)) * 768 + h * 64 + dh];
}

// gather pred output at even slots; split (AD=10, SD=50)
__global__ void gather_kernel(const float* __restrict__ p3, float* __restrict__ out) {
  int i = blockIdx.x * 256 + threadIdx.x;
  if (i >= 245760) return;
  if (i < 40960) {
    int c = i % 10, t = (i / 10) % 256, b = i / 2560;
    out[i] = p3[((size_t)(b * 512 + 2 * t)) * 60 + c];
  } else {
    int j = i - 40960;
    int c = j % 50, t = (j / 50) % 256, b = j / 12800;
    out[i] = p3[((size_t)(b * 512 + 2 * t)) * 60 + 10 + c];
  }
}

// ---------------------------------------------------------------------------
// Host launcher. Input index map assumes depth-first insertion-order
// flattening of the setup_inputs() dict:
//  0 states 1 timesteps 2 actions 3 key_states 4 local_pos 5 global_pos
//  6/7 senc0 W/b  8/9 senc1 W/b  10/11 aenc0 W/b  12/13 aenc1 W/b
//  14..19 pred W/b x3  20/21 ln_f g/b
//  block i base = 22+16*i: ln1g ln1b ln2g ln2b Wq bq Wk bk Wv bv Wo bo W1 b1 W2 b2
// ---------------------------------------------------------------------------
extern "C" void kernel_launch(void* const* d_in, const int* in_sizes, int n_in,
                              void* d_out, int out_size, void* d_ws, size_t ws_size,
                              hipStream_t stream) {
  (void)in_sizes; (void)n_in; (void)out_size; (void)ws_size;
  char* ws = (char*)d_ws;
  const size_t SZ_X = 8192ull * 768 * 4;     // 25,165,824
  const size_t SZ_B = 8192ull * 768 * 2;     // 12,582,912
  size_t off = 0;
  float* X0 = (float*)(ws + off); off += SZ_X;
  float* X1 = (float*)(ws + off); off += SZ_X;
  u16* LNb = (u16*)(ws + off); off += SZ_B;
  u16* QB  = (u16*)(ws + off); off += SZ_B;   // also: states-pad, pred h1
  u16* KB  = (u16*)(ws + off); off += SZ_B;   // also: senc hidden, pred h2
  u16* VB  = (u16*)(ws + off); off += SZ_B;   // also: SE fp32, pred out fp32
  u16* VT  = (u16*)(ws + off); off += SZ_B;   // also: act concat
  u16* Y   = (u16*)(ws + off); off += SZ_B;   // also: aenc hidden
  u16* H1  = (u16*)(ws + off); off += 8192ull * 3072 * 2;  // also: AE fp32
  u16* WB  = (u16*)(ws + off); off += 9437184;

  auto F = [&](int i) { return (const float*)d_in[i]; };
  dim3 B256(256);
  auto wconv = [&](const float* W, int fi, int fo, int Kp, int Np) {
    int tot = Np * Kp;
    wconv_kernel<<<(tot + 255) / 256, B256, 0, stream>>>(W, WB, fi, fo, Kp, Np);
  };

  // ---------------- encoders ----------------
  pad_states_kernel<<<(4096 * 64 + 255) / 256, B256, 0, stream>>>(F(0), QB);
  wconv(F(6), 50, 256, 64, 256);
  gemm_wmma<1, true><<<dim3(256 / 64, (4096 + 127) / 128), B256, 0, stream>>>(
      QB, 64, WB, 64, 256, F(7), nullptr, KB, 256, 4096, 256, 64);
  wconv(F(8), 256, 768, 256, 768);
  gemm_wmma<0, false><<<dim3(768 / 64, (4096 + 127) / 128), B256, 0, stream>>>(
      KB, 256, WB, 256, 768, F(9), nullptr, (float*)VB, 768, 4096, 768, 256);

  concat_act_kernel<<<(4080 * 64 + 255) / 256, B256, 0, stream>>>(F(2), F(3), VT);
  wconv(F(10), 60, 256, 64, 256);
  gemm_wmma<1, true><<<dim3(256 / 64, (4080 + 127) / 128), B256, 0, stream>>>(
      VT, 64, WB, 64, 256, F(11), nullptr, Y, 256, 4080, 256, 64);
  wconv(F(12), 256, 768, 256, 768);
  gemm_wmma<0, false><<<dim3(768 / 64, (4080 + 127) / 128), B256, 0, stream>>>(
      Y, 256, WB, 256, 768, F(13), nullptr, (float*)H1, 768, 4080, 768, 256);

  assemble_kernel<<<(8192 * 768 + 255) / 256, B256, 0, stream>>>(
      (const float*)VB, (const float*)H1, F(5), F(4), (const int*)d_in[1], X0);

  // ---------------- transformer layers ----------------
  dim3 gD(768 / 64, 8192 / 128);
  for (int L = 0; L < 6; L++) {
    int p = 22 + 16 * L;
    // x_ln = LN1(x)
    ln_kernel<<<8192 / 8, B256, 0, stream>>>(X0, F(p + 0), F(p + 1), LNb, 8192);
    // Q, K, V projections
    wconv(F(p + 4), 768, 768, 768, 768);
    gemm_wmma<0, true><<<gD, B256, 0, stream>>>(LNb, 768, WB, 768, 768, F(p + 5),
                                                nullptr, QB, 768, 8192, 768, 768);
    wconv(F(p + 6), 768, 768, 768, 768);
    gemm_wmma<0, true><<<gD, B256, 0, stream>>>(LNb, 768, WB, 768, 768, F(p + 7),
                                                nullptr, KB, 768, 8192, 768, 768);
    wconv(F(p + 8), 768, 768, 768, 768);
    gemm_wmma<0, true><<<gD, B256, 0, stream>>>(LNb, 768, WB, 768, 768, F(p + 9),
                                                nullptr, VB, 768, 8192, 768, 768);
    vtrans_kernel<<<(192 * 64 * 512 + 255) / 256, B256, 0, stream>>>(VB, VT);
    // y = softmax(QK^T/8, causal) V   (flash, WMMA)
    attn_flash<<<dim3(4, 192), B256, 0, stream>>>(QB, KB, VT, Y);
    // x1 = x + y Wo + bo
    wconv(F(p + 10), 768, 768, 768, 768);
    gemm_wmma<0, false><<<gD, B256, 0, stream>>>(Y, 768, WB, 768, 768, F(p + 11),
                                                 X0, X1, 768, 8192, 768, 768);
    // h = GELU(LN2(x1) W1 + b1)
    ln_kernel<<<8192 / 8, B256, 0, stream>>>(X1, F(p + 2), F(p + 3), LNb, 8192);
    wconv(F(p + 12), 768, 3072, 768, 3072);
    gemm_wmma<2, true><<<dim3(3072 / 64, 8192 / 128), B256, 0, stream>>>(
        LNb, 768, WB, 768, 3072, F(p + 13), nullptr, H1, 3072, 8192, 3072, 768);
    // x = x1 + h W2 + b2
    wconv(F(p + 14), 3072, 768, 3072, 768);
    gemm_wmma<0, false><<<gD, B256, 0, stream>>>(H1, 3072, WB, 3072, 768, F(p + 15),
                                                 X1, X0, 768, 8192, 768, 3072);
  }

  // ---------------- head ----------------
  ln_kernel<<<8192 / 8, B256, 0, stream>>>(X0, F(20), F(21), LNb, 8192);
  wconv(F(14), 768, 256, 768, 256);
  gemm_wmma<1, true><<<dim3(256 / 64, 8192 / 128), B256, 0, stream>>>(
      LNb, 768, WB, 768, 256, F(15), nullptr, QB, 256, 8192, 256, 768);
  wconv(F(16), 256, 256, 256, 256);
  gemm_wmma<1, true><<<dim3(256 / 64, 8192 / 128), B256, 0, stream>>>(
      QB, 256, WB, 256, 256, F(17), nullptr, KB, 256, 8192, 256, 256);
  wconv(F(18), 256, 60, 256, 64);
  gemm_wmma<0, false><<<dim3(1, 8192 / 128), B256, 0, stream>>>(
      KB, 256, WB, 256, 64, F(19), nullptr, (float*)VB, 60, 8192, 60, 256);
  gather_kernel<<<(245760 + 255) / 256, B256, 0, stream>>>((const float*)VB,
                                                           (float*)d_out);
}
